// QueryDecoder_35966056136794
// MI455X (gfx1250) — matrix operand order repaired and checked by reading, not compiled, not run on hardware
//
#include <hip/hip_runtime.h>
#include <hip/hip_bf16.h>
#include <math.h>

typedef __attribute__((ext_vector_type(16))) __bf16 v16bf;
typedef __attribute__((ext_vector_type(8)))  float  v8f;

#define NPTS    65536
#define HV      176
#define WV      176
#define DV      132
#define FDIM    32
#define LATD    128
#define OUTD    257
#define OUTPAD  272

// ---- workspace layout (bytes) ----
#define WS_SCALAR   0            // float total_abs
#define WS_FRAGW    256          // 8 layers * 128*128 bf16 (B-fragment order)
#define WS_FRAGW9   (256 + 262144)          // 17*4*32*16 bf16
#define WS_BCAT     (262400 + 69632)        // 1024 + 272 floats (b1..b8, b9 padded)
#define WS_BIASRAW  337408                  // N*128 floats

// softplus(beta=100, threshold=20): for x <= 0.2, y = 100*min(x,0.2) <= 20,
// 1+e^y in [1, 4.9e8] -> no cancellation; hardware exp/log is fine.
// Branchless select keeps EXEC uniform in the WMMA epilogue.
__device__ __forceinline__ float softplus100(float x) {
    float y = fminf(x, 0.2f) * 100.0f;
    float s = __logf(1.0f + __expf(y)) * 0.01f;
    return (x > 0.2f) ? x : s;
}

// ---------------------------------------------------------------- init scalar
__global__ void zero_scalar_kernel(float* s) { *s = 0.0f; }

// ---------------------------------------------------------------- weight prep
struct PrepPtrs {
    const float* W[8];   // W1..W8 (W4 has 125 rows)
    const float* W9;     // 257x128
    const float* bv[9];  // b1..b8, b9
};

// B-fragment layout for v_wmma_f32_16x16x32_bf16, B = 32x16 (K x N):
//   lane: n = lane&15, K-group = (lane>>4)*16 ; element i: k = kchunk*32 + group + i
// frag array index: ((ntile*4 + kchunk)*32 + lane)*16 + i  -> 32B contiguous per lane
__global__ void prep_weights_kernel(PrepPtrs p, __bf16* fragW, __bf16* fragW9, float* bcat) {
    int t = blockIdx.x * blockDim.x + threadIdx.x;
    if (t < 131072) {                       // layers 1..8
        int layer = t >> 14;
        int r     = t & 16383;
        int ntile = r >> 11;
        int kch   = (r >> 9) & 3;
        int lane  = (r >> 4) & 31;
        int i     = r & 15;
        int n = ntile * 16 + (lane & 15);
        int k = kch * 32 + (lane >> 4) * 16 + i;
        float v = (layer == 3 && n >= 125) ? 0.0f : p.W[layer][n * LATD + k];
        fragW[t] = (__bf16)v;
    } else if (t < 165888) {                // layer 9 (padded to 272 rows)
        int r     = t - 131072;
        int ntile = r >> 11;
        int kch   = (r >> 9) & 3;
        int lane  = (r >> 4) & 31;
        int i     = r & 15;
        int n = ntile * 16 + (lane & 15);
        int k = kch * 32 + (lane >> 4) * 16 + i;
        float v = (n < OUTD) ? p.W9[n * LATD + k] : 0.0f;
        fragW9[r] = (__bf16)v;
    } else if (t < 166912) {                // b1..b8
        int idx = t - 165888;
        bcat[idx] = p.bv[idx >> 7][idx & 127];
    } else if (t < 167184) {                // b9 padded
        int c = t - 166912;
        bcat[1024 + c] = (c < OUTD) ? p.bv[8][c] : 0.0f;
    }
}

// ---------------------------------------------------------------- query/bias
// one wave per point; lane = feature channel (FDIM==32==wave32)
__global__ __launch_bounds__(256) void bias_kernel(
    const float* __restrict__ pixel, const float* __restrict__ vol,
    const float* __restrict__ Wq, const float* __restrict__ bq,
    float* __restrict__ biasraw, float* __restrict__ total_abs)
{
    __shared__ float featlds[8][FDIM];
    int wv   = threadIdx.x >> 5;
    int lane = threadIdx.x & 31;
    int pt   = blockIdx.x * 8 + wv;

    float qx = pixel[pt * 3 + 0] + 24.0f;
    float qy = pixel[pt * 3 + 1] + 24.0f;
    float qz = (pixel[pt * 3 + 2] - 425.0f) * (128.0f / 480.0f);
    int bx = (int)floorf(qx), by = (int)floorf(qy), bz = (int)floorf(qz);

    // feat = M / sum(M), M_f = sum_k sim_k * vol[idx_k][f]  (normalizations collapse)
    float Mf = 0.0f;
    for (int dx = -1; dx <= 2; ++dx) {
        float kx = (float)(bx + dx);
        int ox = (bx + dx) * WV;
        for (int dy = -1; dy <= 2; ++dy) {
            float ky = (float)(by + dy);
            int oy = (ox + by + dy) * DV;
            #pragma unroll
            for (int dz = -1; dz <= 2; ++dz) {
                float kz  = (float)(bz + dz);
                float sim = kx * qx + ky * qy + kz * qz;
                int idx = (oy + bz + dz) * FDIM + lane;   // lane-contiguous gather
                Mf += sim * vol[idx];
            }
        }
    }
    float tot = Mf;
    #pragma unroll
    for (int off = 16; off; off >>= 1) tot += __shfl_xor(tot, off, 32);
    featlds[wv][lane] = Mf / tot;

    // biasraw = feat @ Wq.T + bq  (each lane: 4 outputs)
    float absacc = 0.0f;
    #pragma unroll
    for (int cc = 0; cc < 4; ++cc) {
        int o = cc * 32 + lane;
        float acc = bq[o];
        #pragma unroll
        for (int f = 0; f < FDIM; ++f) acc += featlds[wv][f] * Wq[o * FDIM + f];
        biasraw[pt * LATD + o] = acc;
        absacc += fabsf(acc);
    }
    #pragma unroll
    for (int off = 16; off; off >>= 1) absacc += __shfl_xor(absacc, off, 32);
    if (lane == 0) atomicAdd(total_abs, absacc);
}

// ---------------------------------------------------------------- WMMA MLP
// block = 128 threads = 4 waves; each wave owns a 16-point tile
#define MLPWAVES 4
__global__ __launch_bounds__(128) void mlp_kernel(
    const float* __restrict__ world, const float* __restrict__ biasraw,
    const float* __restrict__ total_abs,
    const __bf16* __restrict__ fragW, const __bf16* __restrict__ fragW9,
    const float* __restrict__ bcat,
    const float* __restrict__ W0, const float* __restrict__ b0,
    float* __restrict__ out)
{
    __shared__ __bf16 wlds[18432];                 // 36864 B weight stage
    __shared__ __bf16 act[MLPWAVES][16][LATD];     // 16384 B activations (bf16)

    int tid  = threadIdx.x;
    int wv   = tid >> 5;
    int lane = tid & 31;
    int m16  = lane & 15;
    int hl   = lane >> 4;
    int pt0  = (blockIdx.x * MLPWAVES + wv) * 16;

    float scale = 8388608.0f / total_abs[0];       // N*128 / sum|bias|

    // ---- layer 0: 3 -> 128, * bias, softplus (VALU) ----
    #pragma unroll
    for (int cc = 0; cc < 4; ++cc) {
        int c = cc * 32 + lane;
        float w0x = W0[c * 3 + 0], w0y = W0[c * 3 + 1], w0z = W0[c * 3 + 2], bb = b0[c];
        for (int m = 0; m < 16; ++m) {
            int p = pt0 + m;
            float x = w0x * world[p * 3 + 0] + w0y * world[p * 3 + 1]
                    + w0z * world[p * 3 + 2] + bb;
            x *= biasraw[p * LATD + c] * scale;
            act[wv][m][c] = (__bf16)softplus100(x);
        }
    }

    // ---- layers 1..8: 128->128 WMMA bf16/f32 ----
    for (int gl = 1; gl <= 8; ++gl) {
        __syncthreads();
        const uint4* src = (const uint4*)(fragW + (gl - 1) * 16384);
        for (int t = tid; t < 2048; t += 128) ((uint4*)wlds)[t] = src[t];
        __syncthreads();

        // A fragments: 16x32 bf16 each, element s: k = kc*32 + hl*16 + s (same order as B)
        v16bf afrag[4];
        #pragma unroll
        for (int kc = 0; kc < 4; ++kc) {
            #pragma unroll
            for (int j = 0; j < 8; ++j) {
                int kb = kc * 32 + hl * 16 + 2 * j;
                afrag[kc][2 * j]     = act[wv][m16][kb];
                afrag[kc][2 * j + 1] = act[wv][m16][kb + 1];
            }
        }
        const float* bl = bcat + (gl - 1) * LATD;
        #pragma unroll
        for (int nt = 0; nt < 8; ++nt) {
            int c = nt * 16 + m16;
            float bv = bl[c];
            v8f acc = {bv, bv, bv, bv, bv, bv, bv, bv};
            #pragma unroll
            for (int kc = 0; kc < 4; ++kc) {
                v16bf bfrag = *(const v16bf*)&wlds[((nt * 4 + kc) * 32 + lane) * 16];
                acc = __builtin_amdgcn_wmma_f32_16x16x32_bf16(
                          false, afrag[kc], false, bfrag, (short)0, acc, false, false);
            }
            #pragma unroll
            for (int r = 0; r < 8; ++r) {
                int m = r + 8 * hl;
                float x = acc[r];
                if (gl == 4 && c >= 125)               // skip-concat of world coords
                    x = world[(pt0 + m) * 3 + (c - 125)];
                act[wv][m][c] = (__bf16)softplus100(x); // softplus after concat (ref order)
            }
        }
    }

    // ---- layer 9: 128 -> 257 (padded 272), two LDS weight chunks ----
    v16bf afrag[4];
    #pragma unroll
    for (int kc = 0; kc < 4; ++kc) {
        #pragma unroll
        for (int j = 0; j < 8; ++j) {
            int kb = kc * 32 + hl * 16 + 2 * j;
            afrag[kc][2 * j]     = act[wv][m16][kb];
            afrag[kc][2 * j + 1] = act[wv][m16][kb + 1];
        }
    }
    const float* b9 = bcat + 1024;
    // chunk 0: ntiles 0..8
    __syncthreads();
    for (int t = tid; t < 2304; t += 128) ((uint4*)wlds)[t] = ((const uint4*)fragW9)[t];
    __syncthreads();
    #pragma unroll
    for (int nt = 0; nt < 9; ++nt) {
        int c = nt * 16 + m16;
        float bv = b9[c];
        v8f acc = {bv, bv, bv, bv, bv, bv, bv, bv};
        #pragma unroll
        for (int kc = 0; kc < 4; ++kc) {
            v16bf bfrag = *(const v16bf*)&wlds[((nt * 4 + kc) * 32 + lane) * 16];
            acc = __builtin_amdgcn_wmma_f32_16x16x32_bf16(
                      false, afrag[kc], false, bfrag, (short)0, acc, false, false);
        }
        #pragma unroll
        for (int r = 0; r < 8; ++r)
            out[(pt0 + r + 8 * hl) * OUTD + c] = acc[r];
    }
    // chunk 1: ntiles 9..16
    __syncthreads();
    for (int t = tid; t < 2048; t += 128)
        ((uint4*)wlds)[t] = ((const uint4*)(fragW9 + 18432))[t];
    __syncthreads();
    #pragma unroll
    for (int nt = 9; nt < 17; ++nt) {
        int c = nt * 16 + m16;
        float bv = b9[c];
        v8f acc = {bv, bv, bv, bv, bv, bv, bv, bv};
        int nt2 = nt - 9;
        #pragma unroll
        for (int kc = 0; kc < 4; ++kc) {
            v16bf bfrag = *(const v16bf*)&wlds[((nt2 * 4 + kc) * 32 + lane) * 16];
            acc = __builtin_amdgcn_wmma_f32_16x16x32_bf16(
                      false, afrag[kc], false, bfrag, (short)0, acc, false, false);
        }
        if (c < OUTD) {
            #pragma unroll
            for (int r = 0; r < 8; ++r)
                out[(pt0 + r + 8 * hl) * OUTD + c] = acc[r];
        }
    }
}

// ---------------------------------------------------------------- launch
extern "C" void kernel_launch(void* const* d_in, const int* in_sizes, int n_in,
                              void* d_out, int out_size, void* d_ws, size_t ws_size,
                              hipStream_t stream) {
    const float* world = (const float*)d_in[0];
    const float* pixel = (const float*)d_in[1];
    const float* vol   = (const float*)d_in[2];
    const float* Wq    = (const float*)d_in[3];
    const float* bq    = (const float*)d_in[4];
    const float* W0    = (const float*)d_in[5];
    const float* b0    = (const float*)d_in[6];

    char* ws = (char*)d_ws;
    float*  total_abs = (float*)(ws + WS_SCALAR);
    __bf16* fragW     = (__bf16*)(ws + WS_FRAGW);
    __bf16* fragW9    = (__bf16*)(ws + WS_FRAGW9);
    float*  bcat      = (float*)(ws + WS_BCAT);
    float*  biasraw   = (float*)(ws + WS_BIASRAW);
    float*  out       = (float*)d_out;

    PrepPtrs pp;
    for (int l = 0; l < 8; ++l) pp.W[l] = (const float*)d_in[7 + 2 * l];   // W1..W8
    pp.W9 = (const float*)d_in[23];
    for (int l = 0; l < 8; ++l) pp.bv[l] = (const float*)d_in[8 + 2 * l];  // b1..b8
    pp.bv[8] = (const float*)d_in[24];                                     // b9

    zero_scalar_kernel<<<1, 1, 0, stream>>>(total_abs);
    prep_weights_kernel<<<654, 256, 0, stream>>>(pp, fragW, fragW9, bcat);
    bias_kernel<<<NPTS / 8, 256, 0, stream>>>(pixel, vol, Wq, bq, biasraw, total_abs);
    mlp_kernel<<<NPTS / (MLPWAVES * 16), 128, 0, stream>>>(
        world, biasraw, total_abs, fragW, fragW9, bcat, W0, b0, out);
}
